// Net_86268713108140
// MI455X (gfx1250) — compile-verified, matrix operand order checked by reading, not run on hardware
//
#include <hip/hip_runtime.h>
#include <hip/hip_bf16.h>
#include <math.h>
#include <stdint.h>

// ---------------------------------------------------------------------------
// Sizes (fixed by the reference)
// ---------------------------------------------------------------------------
#define N_NODES 20000
#define E_EDGES 640000
#define CCH     128     // hidden channels
#define HH      8       // heads
#define DDIM    16      // per-head dim
#define GG      50      // gaussians
#define GPAD    64      // padded K for edge_attr GEMM
#define FILN    128     // filters
#define LL      3       // interaction blocks
#define CUTF    10.0f
#define LOG2F_  0.6931471805599453f

typedef _Float16 half_t;
typedef __attribute__((ext_vector_type(16))) _Float16 v16h;
typedef __attribute__((ext_vector_type(8)))  _Float16 v8h;
typedef __attribute__((ext_vector_type(8)))  float    v8f;

__device__ __forceinline__ float sspf(float x) {
    // shifted softplus: softplus(x) - ln(2), numerically stable
    return fmaxf(x, 0.0f) + log1pf(__expf(-fabsf(x))) - LOG2F_;
}

// float atomic max via sign-magnitude int trick (init value is -inf)
__device__ __forceinline__ void atomicMaxF(float* addr, float v) {
    if (v >= 0.0f) atomicMax((int*)addr, __float_as_int(v));
    else           atomicMin((unsigned int*)addr, __float_as_uint(v));
}

// ---------------------------------------------------------------------------
// WMMA fragment loader. Layout per CDNA5 ISA (16-bit A 16x32, wave32):
//   lanes 0-15 : row r=lane,    K = 0..7  and 16..23
//   lanes 16-31: row r=lane-16, K = 8..15 and 24..31
// B mirrors with "row" = output column (weights packed transposed, K
// contiguous), so one loader serves both operands. Works for global or LDS
// pointers (compiler infers address space -> global_load_b128 / ds_load_b128).
// ---------------------------------------------------------------------------
__device__ __forceinline__ v16h load_frag(const half_t* p, int ld) {
    const int lane = threadIdx.x & 31;
    const int r    = lane & 15;
    const int klo  = (lane & 16) >> 1;   // 0 or 8
    const half_t* base = p + (size_t)r * ld + klo;
    v8h lo = *(const v8h*)(base);
    v8h hi = *(const v8h*)(base + 16);
    v16h out;
#pragma unroll
    for (int i = 0; i < 8; ++i) { out[i] = lo[i]; out[i + 8] = hi[i]; }
    return out;
}

#define WMMA_F16(a, b, c) \
    __builtin_amdgcn_wmma_f32_16x16x32_f16(false, (a), false, (b), (short)0, (c), false, false)

// ---------------------------------------------------------------------------
// Generic f16 GEMM, f32 accumulate:  out[M,Ncols] = A[M,K] @ W[K,Ncols] + bias
// One wave per 16x16 output tile; block = (Ncols/16) waves covers 16 rows x
// all columns. Optional residual add, ShiftedSoftplus, dual f32/f16 stores.
// ---------------------------------------------------------------------------
__global__ void gemm16_kernel(const half_t* __restrict__ A, int lda,
                              const half_t* __restrict__ Wt, int K,
                              const float*  __restrict__ bias,
                              const float*  __restrict__ addsrc,
                              float*        __restrict__ outf,
                              half_t*       __restrict__ outh,
                              int Ncols, int act) {
    const int row0 = blockIdx.x << 4;
    const int wave = threadIdx.x >> 5;
    const int n0   = wave << 4;
    const int lane = threadIdx.x & 31;

    const half_t* Arow = A  + (size_t)row0 * lda;
    const half_t* Wn   = Wt + (size_t)n0 * K;

    v8f acc = {};
    for (int kb = 0; kb < K; kb += 32) {
        __builtin_prefetch(Arow + kb + 32, 0, 3);
        v16h a = load_frag(Arow + kb, lda);
        v16h b = load_frag(Wn + kb, K);
        acc = WMMA_F16(a, b, acc);
    }

    // D layout: VGPR r -> (M=r, N=lane) lanes 0-15, (M=r+8, N=lane-16) 16-31
    const int ncol  = lane & 15;
    const int mbase = (lane & 16) >> 1;  // 0 or 8
    const int gn    = n0 + ncol;
    const float bv  = bias ? bias[gn] : 0.0f;
#pragma unroll
    for (int r = 0; r < 8; ++r) {
        const int gm = row0 + mbase + r;
        const size_t idx = (size_t)gm * Ncols + gn;
        float v = acc[r] + bv;
        if (addsrc) v += addsrc[idx];
        if (act)    v  = sspf(v);
        if (outf)   outf[idx] = v;
        if (outh)   outh[idx] = (half_t)v;
    }
}

// ---------------------------------------------------------------------------
// Fused edge-filter MLP + edge embedding (3 chained WMMA GEMMs, LDS staged):
//   ep = (ssp(EA @ W1 + b1) @ W2 + b2) @ We + be          -> EP16 [E, C] f16
// One block = 16 edges, 8 waves cover all 128 output columns.
// The 16x64 f16 edge_attr tile is one contiguous 2KB block; it is fetched
// ONCE per block into LDS with async copy (global_load_async_to_lds_b128,
// tracked by ASYNCcnt) instead of 8 redundant per-wave global loads.
// Inter-stage transpose (D-layout -> A-layout) goes through a 16x136-half
// LDS tile (padded stride keeps v8h reads 16B aligned).
// ---------------------------------------------------------------------------
#define LDW 136
__global__ void edge_mlp_fused_kernel(const half_t* __restrict__ EA,
                                      const half_t* __restrict__ w1t,
                                      const float*  __restrict__ b1,
                                      const half_t* __restrict__ w2t,
                                      const float*  __restrict__ b2,
                                      const half_t* __restrict__ wet,
                                      const float*  __restrict__ bev,
                                      half_t* __restrict__ EP16) {
    __shared__ __align__(16) half_t tile[16 * LDW];
    __shared__ __align__(16) half_t eatile[16 * GPAD];   // 2KB A tile

    const int row0  = blockIdx.x << 4;
    const int wave  = threadIdx.x >> 5;
    const int n0    = wave << 4;
    const int lane  = threadIdx.x & 31;
    const int ncol  = lane & 15;
    const int mbase = (lane & 16) >> 1;
    const int gn    = n0 + ncol;

    // ---- async copy of the contiguous 16x64 f16 A tile into LDS -----------
    // 2048 bytes = 128 lanes x 16B. LDS dest address = low 32 bits of the
    // generic pointer (flat aperture maps ADDR[31:0] to wave-relative LDS).
    if (threadIdx.x < 128) {
        const uint32_t ldsa =
            (uint32_t)(uintptr_t)(&eatile[0]) + threadIdx.x * 16u;
        const half_t* gsrc = EA + (size_t)row0 * GPAD + threadIdx.x * 8u;
        asm volatile("global_load_async_to_lds_b128 %0, %1, off"
                     :: "v"(ldsa), "v"(gsrc) : "memory");
    }
    asm volatile("s_wait_asynccnt 0" ::: "memory");
    __syncthreads();

    // ---- stage 1: t = ssp(EA @ W1 + b1), K = 64, A from LDS ---------------
    v8f acc = {};
    {
        const half_t* Wn = w1t + (size_t)n0 * GPAD;
#pragma unroll
        for (int kb = 0; kb < GPAD; kb += 32) {
            v16h a = load_frag(eatile + kb, GPAD);
            v16h b = load_frag(Wn + kb, GPAD);
            acc = WMMA_F16(a, b, acc);
        }
        const float bvv = b1[gn];
#pragma unroll
        for (int r = 0; r < 8; ++r)
            tile[(mbase + r) * LDW + gn] = (half_t)sspf(acc[r] + bvv);
    }
    __syncthreads();

    // ---- stage 2: e = t @ W2 + b2, K = 128, A from LDS --------------------
    acc = v8f{};
    {
        const half_t* Wn = w2t + (size_t)n0 * CCH;
#pragma unroll
        for (int kb = 0; kb < CCH; kb += 32) {
            v16h a = load_frag(tile + kb, LDW);
            v16h b = load_frag(Wn + kb, CCH);
            acc = WMMA_F16(a, b, acc);
        }
    }
    __syncthreads();   // all waves done reading tile before overwrite
    {
        const float bvv = b2[gn];
#pragma unroll
        for (int r = 0; r < 8; ++r)
            tile[(mbase + r) * LDW + gn] = (half_t)(acc[r] + bvv);
    }
    __syncthreads();

    // ---- stage 3: ep = e @ We + be, K = 128, A from LDS -------------------
    acc = v8f{};
    {
        const half_t* Wn = wet + (size_t)n0 * CCH;
#pragma unroll
        for (int kb = 0; kb < CCH; kb += 32) {
            v16h a = load_frag(tile + kb, LDW);
            v16h b = load_frag(Wn + kb, CCH);
            acc = WMMA_F16(a, b, acc);
        }
        const float bvv = bev[gn];
#pragma unroll
        for (int r = 0; r < 8; ++r) {
            const int gm = row0 + mbase + r;
            EP16[(size_t)gm * CCH + gn] = (half_t)(acc[r] + bvv);
        }
    }
}

// ---------------------------------------------------------------------------
// Weight pack: Wt[n, k] = f16(W[k, n]), zero-padded to Kpad
// ---------------------------------------------------------------------------
__global__ void pack_wt_kernel(const float* __restrict__ W, half_t* __restrict__ Wt,
                               int Kin, int Kpad, int Ncols) {
    const int i = blockIdx.x * blockDim.x + threadIdx.x;
    if (i >= Kpad * Ncols) return;
    const int n = i / Kpad, k = i - n * Kpad;
    Wt[(size_t)n * Kpad + k] = (k < Kin) ? (half_t)W[(size_t)k * Ncols + n]
                                         : (half_t)0.0f;
}

// ---------------------------------------------------------------------------
// Edge geometry + gaussian smearing -> f16 [E, GPAD] (pad zero)
// ---------------------------------------------------------------------------
__global__ void edge_geom_kernel(const int* __restrict__ ei,
                                 const float* __restrict__ pos,
                                 half_t* __restrict__ ea) {
    const int e = blockIdx.x * blockDim.x + threadIdx.x;
    if (e >= E_EDGES) return;
    const int s = ei[e];
    const int t = ei[E_EDGES + e];
    const float dx = pos[t * 3 + 0] - pos[s * 3 + 0];
    const float dy = pos[t * 3 + 1] - pos[s * 3 + 1];
    const float dz = pos[t * 3 + 2] - pos[s * 3 + 2];
    const float d  = sqrtf(dx * dx + dy * dy + dz * dz + 1e-12f);
    const float step  = CUTF / (float)(GG - 1);
    const float coeff = -0.5f / (step * step);
    half_t* row = ea + (size_t)e * GPAD;
    for (int g = 0; g < GG; ++g) {
        const float diff = d - (float)g * step;
        row[g] = (half_t)__expf(coeff * diff * diff);
    }
    for (int g = GG; g < GPAD; ++g) row[g] = (half_t)0.0f;
}

// ---------------------------------------------------------------------------
// Node embedding gather -> f32 H and f16 H16
// ---------------------------------------------------------------------------
__global__ void embed_kernel(const int* __restrict__ z, const float* __restrict__ emb,
                             float* __restrict__ H, half_t* __restrict__ H16) {
    const int i = blockIdx.x * blockDim.x + threadIdx.x;
    if (i >= N_NODES * CCH) return;
    const int n = i >> 7, c = i & 127;
    const float v = emb[(size_t)z[n] * CCH + c];
    H[i] = v;
    H16[i] = (half_t)v;
}

// ---------------------------------------------------------------------------
// Attention helpers (segment softmax via atomics)
// ---------------------------------------------------------------------------
__global__ void attn_init_kernel(float* maxb, float* den, float* agg) {
    const int i = blockIdx.x * blockDim.x + threadIdx.x;
    if (i < N_NODES * HH) { maxb[i] = -INFINITY; den[i] = 0.0f; }
    if (i < N_NODES * CCH) agg[i] = 0.0f;
}

__global__ void attn_logits_kernel(const int* __restrict__ ei,
                                   const float* __restrict__ Q,
                                   const float* __restrict__ Kn,
                                   const half_t* __restrict__ EP,
                                   float* __restrict__ logits,
                                   float* __restrict__ maxb) {
    const int i = blockIdx.x * blockDim.x + threadIdx.x;
    if (i >= E_EDGES * HH) return;
    const int e = i >> 3, h = i & 7;
    const int s = ei[e];
    const int t = ei[E_EDGES + e];
    const float*  qp = Q  + (size_t)t * CCH + h * DDIM;
    const float*  kp = Kn + (size_t)s * CCH + h * DDIM;
    const half_t* ep = EP + (size_t)e * CCH + h * DDIM;
    float acc = 0.0f;
#pragma unroll
    for (int d0 = 0; d0 < DDIM; ++d0) acc += qp[d0] * (kp[d0] + (float)ep[d0]);
    acc *= 0.25f;  // 1/sqrt(D)
    logits[i] = acc;
    atomicMaxF(maxb + (size_t)t * HH + h, acc);
}

__global__ void attn_exp_kernel(const int* __restrict__ ei,
                                float* __restrict__ logits,
                                const float* __restrict__ maxb,
                                float* __restrict__ den) {
    const int i = blockIdx.x * blockDim.x + threadIdx.x;
    if (i >= E_EDGES * HH) return;
    const int e = i >> 3, h = i & 7;
    const int t = ei[E_EDGES + e];
    float m = maxb[(size_t)t * HH + h];
    if (m < -1e37f) m = 0.0f;  // segment_max of empty segment -> 0
    const float ex = __expf(logits[i] - m);
    logits[i] = ex;  // overwrite logits with exp numerator
    atomicAdd(den + (size_t)t * HH + h, ex);
}

__global__ void attn_agg_kernel(const int* __restrict__ ei,
                                const float* __restrict__ exv,
                                const float* __restrict__ den,
                                const float* __restrict__ Vn,
                                const half_t* __restrict__ EP,
                                float* __restrict__ agg) {
    const int i = blockIdx.x * blockDim.x + threadIdx.x;
    if (i >= E_EDGES * CCH) return;
    const int e = i >> 7, c = i & 127, h = c >> 4;
    const int s = ei[e];
    const int t = ei[E_EDGES + e];
    const float alpha = exv[(size_t)e * HH + h] /
                        (den[(size_t)t * HH + h] + 1e-16f);
    const float vv = Vn[(size_t)s * CCH + c] + (float)EP[(size_t)e * CCH + c];
    atomicAdd(agg + (size_t)t * CCH + c, alpha * vv);
}

// ---------------------------------------------------------------------------
// Final projection: out[n] = O1[n,:64] . w2 + b2
// ---------------------------------------------------------------------------
__global__ void head2_kernel(const half_t* __restrict__ O1,
                             const float* __restrict__ w2,
                             const float* __restrict__ b2,
                             float* __restrict__ out) {
    const int n = blockIdx.x * blockDim.x + threadIdx.x;
    if (n >= N_NODES) return;
    float acc = b2[0];
    const half_t* row = O1 + (size_t)n * (CCH / 2);
#pragma unroll
    for (int j = 0; j < CCH / 2; ++j) acc += (float)row[j] * w2[j];
    out[n] = acc;
}

// ---------------------------------------------------------------------------
// Host-side orchestration
// ---------------------------------------------------------------------------
static inline unsigned cdiv(size_t a, unsigned b) { return (unsigned)((a + b - 1) / b); }

extern "C" void kernel_launch(void* const* d_in, const int* in_sizes, int n_in,
                              void* d_out, int out_size, void* d_ws, size_t ws_size,
                              hipStream_t stream) {
    (void)in_sizes; (void)n_in; (void)out_size; (void)ws_size;

    const int*   z      = (const int*)  d_in[0];
    const float* pos    = (const float*)d_in[1];
    const int*   ei     = (const int*)  d_in[2];
    const float* emb    = (const float*)d_in[3];
    const float* mlp_w1 = (const float*)d_in[4];
    const float* mlp_b1 = (const float*)d_in[5];
    const float* mlp_w2 = (const float*)d_in[6];
    const float* mlp_b2 = (const float*)d_in[7];
    const float* wq     = (const float*)d_in[8];
    const float* bq     = (const float*)d_in[9];
    const float* wk     = (const float*)d_in[10];
    const float* bk     = (const float*)d_in[11];
    const float* wv     = (const float*)d_in[12];
    const float* bv     = (const float*)d_in[13];
    const float* we     = (const float*)d_in[14];
    const float* be     = (const float*)d_in[15];
    const float* wskip  = (const float*)d_in[16];
    const float* bskip  = (const float*)d_in[17];
    const float* wl     = (const float*)d_in[18];
    const float* bl     = (const float*)d_in[19];
    const float* w_o1   = (const float*)d_in[20];
    const float* b_o1   = (const float*)d_in[21];
    const float* w_o2   = (const float*)d_in[22];
    const float* b_o2   = (const float*)d_in[23];
    float* outp = (float*)d_out;

    // bump allocator over workspace (256B aligned)
    char* wsB = (char*)d_ws;
    size_t off = 0;
    auto bump = [&](size_t bytes) -> char* {
        off = (off + 255) & ~(size_t)255;
        char* p = wsB + off;
        off += bytes;
        return p;
    };

    half_t* EA   = (half_t*)bump((size_t)E_EDGES * GPAD * sizeof(half_t));
    half_t* EP16 = (half_t*)bump((size_t)E_EDGES * CCH * sizeof(half_t));
    float*  LOGB = (float*) bump((size_t)E_EDGES * HH * sizeof(float));
    float*  MAXB = (float*) bump((size_t)N_NODES * HH * sizeof(float));
    float*  DEN  = (float*) bump((size_t)N_NODES * HH * sizeof(float));
    float*  AGG  = (float*) bump((size_t)N_NODES * CCH * sizeof(float));
    float*  Hf   = (float*) bump((size_t)N_NODES * CCH * sizeof(float));
    half_t* H16  = (half_t*)bump((size_t)N_NODES * CCH * sizeof(half_t));
    float*  Qf   = (float*) bump((size_t)N_NODES * CCH * sizeof(float));
    float*  Knf  = (float*) bump((size_t)N_NODES * CCH * sizeof(float));
    float*  Vnf  = (float*) bump((size_t)N_NODES * CCH * sizeof(float));
    float*  TMP  = (float*) bump((size_t)N_NODES * CCH * sizeof(float));
    half_t* TMP16= (half_t*)bump((size_t)N_NODES * CCH * sizeof(half_t));
    half_t* O1H  = (half_t*)bump((size_t)N_NODES * (CCH / 2) * sizeof(half_t));

    half_t *w1t[LL], *w2t[LL], *wet[LL], *wqt[LL], *wkt[LL], *wvt[LL],
           *wskipt[LL], *wlt[LL];
    for (int l = 0; l < LL; ++l) {
        w1t[l]    = (half_t*)bump((size_t)GPAD * CCH * sizeof(half_t));
        w2t[l]    = (half_t*)bump((size_t)CCH * CCH * sizeof(half_t));
        wet[l]    = (half_t*)bump((size_t)CCH * CCH * sizeof(half_t));
        wqt[l]    = (half_t*)bump((size_t)CCH * CCH * sizeof(half_t));
        wkt[l]    = (half_t*)bump((size_t)CCH * CCH * sizeof(half_t));
        wvt[l]    = (half_t*)bump((size_t)CCH * CCH * sizeof(half_t));
        wskipt[l] = (half_t*)bump((size_t)CCH * CCH * sizeof(half_t));
        wlt[l]    = (half_t*)bump((size_t)CCH * CCH * sizeof(half_t));
    }
    half_t* wo1t = (half_t*)bump((size_t)(CCH / 2) * CCH * sizeof(half_t));

    // ---- weight packing (f32 -> f16, transposed, K-padded) -----------------
    auto pack = [&](const float* W, half_t* Wt, int Kin, int Kpad, int Ncols) {
        const unsigned total = (unsigned)(Kpad * Ncols);
        pack_wt_kernel<<<cdiv(total, 256), 256, 0, stream>>>(W, Wt, Kin, Kpad, Ncols);
    };
    for (int l = 0; l < LL; ++l) {
        pack(mlp_w1 + (size_t)l * GG * FILN, w1t[l], GG, GPAD, FILN);
        pack(mlp_w2 + (size_t)l * FILN * FILN, w2t[l], FILN, FILN, FILN);
        pack(we + (size_t)l * FILN * CCH, wet[l], FILN, FILN, CCH);
        pack(wq + (size_t)l * CCH * CCH, wqt[l], CCH, CCH, CCH);
        pack(wk + (size_t)l * CCH * CCH, wkt[l], CCH, CCH, CCH);
        pack(wv + (size_t)l * CCH * CCH, wvt[l], CCH, CCH, CCH);
        pack(wskip + (size_t)l * CCH * CCH, wskipt[l], CCH, CCH, CCH);
        pack(wl + (size_t)l * CCH * CCH, wlt[l], CCH, CCH, CCH);
    }
    pack(w_o1, wo1t, CCH, CCH, CCH / 2);

    // ---- geometry + embedding ---------------------------------------------
    edge_geom_kernel<<<cdiv(E_EDGES, 256), 256, 0, stream>>>(ei, pos, EA);
    embed_kernel<<<cdiv((size_t)N_NODES * CCH, 256), 256, 0, stream>>>(z, emb, Hf, H16);

    // ---- GEMM launcher -----------------------------------------------------
    auto gemm = [&](const half_t* A, int lda, const half_t* Wt, int K,
                    const float* bias, const float* addsrc,
                    float* outf, half_t* outh, int M, int Ncols, int act) {
        dim3 grid((unsigned)(M / 16));
        dim3 block((unsigned)((Ncols / 16) * 32));
        gemm16_kernel<<<grid, block, 0, stream>>>(A, lda, Wt, K, bias, addsrc,
                                                  outf, outh, Ncols, act);
    };

    // ---- interaction blocks -----------------------------------------------
    for (int l = 0; l < LL; ++l) {
        // fused edge-filter MLP + edge embedding: 10 chained WMMAs per wave,
        // async-to-LDS A-tile fetch
        edge_mlp_fused_kernel<<<E_EDGES / 16, 256, 0, stream>>>(
            EA, w1t[l], mlp_b1 + (size_t)l * FILN,
            w2t[l], mlp_b2 + (size_t)l * FILN,
            wet[l], be + (size_t)l * CCH, EP16);

        // node projections (WMMA)
        gemm(H16, CCH, wqt[l], CCH, bq + (size_t)l * CCH, nullptr,
             Qf, nullptr, N_NODES, CCH, 0);
        gemm(H16, CCH, wkt[l], CCH, bk + (size_t)l * CCH, nullptr,
             Knf, nullptr, N_NODES, CCH, 0);
        gemm(H16, CCH, wvt[l], CCH, bv + (size_t)l * CCH, nullptr,
             Vnf, nullptr, N_NODES, CCH, 0);

        // segment softmax attention (atomics)
        attn_init_kernel<<<cdiv((size_t)N_NODES * CCH, 256), 256, 0, stream>>>(
            MAXB, DEN, AGG);
        attn_logits_kernel<<<cdiv((size_t)E_EDGES * HH, 256), 256, 0, stream>>>(
            ei, Qf, Knf, EP16, LOGB, MAXB);
        attn_exp_kernel<<<cdiv((size_t)E_EDGES * HH, 256), 256, 0, stream>>>(
            ei, LOGB, MAXB, DEN);
        attn_agg_kernel<<<cdiv((size_t)E_EDGES * CCH, 256), 256, 0, stream>>>(
            ei, LOGB, DEN, Vnf, EP16, AGG);

        // out = ssp(agg + h@wskip + bskip); h = h + out@wl + bl
        gemm(H16, CCH, wskipt[l], CCH, bskip + (size_t)l * CCH, AGG,
             TMP, TMP16, N_NODES, CCH, 1);
        gemm(TMP16, CCH, wlt[l], CCH, bl + (size_t)l * CCH, Hf,
             Hf, H16, N_NODES, CCH, 0);
    }

    // ---- output head -------------------------------------------------------
    gemm(H16, CCH, wo1t, CCH, b_o1, nullptr, nullptr, O1H, N_NODES, CCH / 2, 1);
    head2_kernel<<<cdiv(N_NODES, 256), 256, 0, stream>>>(O1H, w_o2, b_o2, outp);
}
